// NA_Spatial_MSA_85263690760543
// MI455X (gfx1250) — matrix operand order repaired and checked by reading, not compiled, not run on hardware
//
#include <hip/hip_runtime.h>

// ---------------------------------------------------------------------------
// CDNA5 wave32 WMMA types
// ---------------------------------------------------------------------------
typedef __attribute__((ext_vector_type(16))) __bf16   bf16x16;
typedef __attribute__((ext_vector_type(2)))  __bf16   bf16x2;
typedef __attribute__((ext_vector_type(8)))  float    f32x8;
typedef __attribute__((ext_vector_type(4)))  unsigned u32x4;

#define Hdim 256
#define Wdim 256
#define Cdim 256
#define WC   (Wdim * Cdim)   // 65536
#define SEQ  64
#define NW   4096

// ---------------------------------------------------------------------------
// float pair -> packed bf16x2 (hardware v_cvt_pk_bf16_f32 when available)
// ---------------------------------------------------------------------------
#if __has_builtin(__builtin_amdgcn_cvt_pk_bf16_f32)
static __device__ inline unsigned pk2(float a, float b) {
    return __builtin_bit_cast(unsigned, __builtin_amdgcn_cvt_pk_bf16_f32(a, b));
}
#else
static __device__ inline unsigned pk2(float a, float b) {
    bf16x2 v; v[0] = (__bf16)a; v[1] = (__bf16)b;
    return __builtin_bit_cast(unsigned, v);
}
#endif

static __device__ inline __bf16 f2bf(float f) {
    unsigned short s = (unsigned short)pk2(f, f);
    return __builtin_bit_cast(__bf16, s);
}

static __device__ inline f32x8 wmma_bf16(bf16x16 a, bf16x16 b, f32x8 c) {
    // (neg_a, A, neg_b, B, c_mod, C, reuse_a, reuse_b)
    return __builtin_amdgcn_wmma_f32_16x16x32_bf16(
        false, a, false, b, (short)0, c, false, false);
}

// A-fragment layout helper: column/K index `kin` (0..31) within a 32-wide K
// block maps to lane-half hr and element idx in the contiguous order used here:
//   K = 16*hr + idx          for idx 0..15
//   => hr = kin>>4 ; idx = kin&15

// ---------------------------------------------------------------------------
// Prep: Wf = Wv @ Wo (fused linear), stored pre-swizzled in B-fragment order:
// one contiguous 32-byte chunk per (ktile, ntile, lane).
//   B 32x16 layout: lane = N + 16*h, element idx -> K = 16*h + idx
// ---------------------------------------------------------------------------
__global__ __launch_bounds__(256)
void wf_prep_kernel(const float* __restrict__ Wv,
                    const float* __restrict__ Wo,
                    void* __restrict__ wf_raw) {
    __bf16* wf = (__bf16*)wf_raw;
    const int a = blockIdx.x;      // K index (input dim of fused linear)
    const int b = threadIdx.x;     // N index (output channel)
    float acc = 0.f;
#pragma unroll 8
    for (int t = 0; t < 256; ++t)
        acc = fmaf(Wv[a * 256 + t], Wo[t * 256 + b], acc);
    const int kt = a >> 5, kin = a & 31;
    const int nt = b >> 4, cc = b & 15;
    const int hh = kin >> 4, idx = kin & 15;
    const int lane = cc + 16 * hh;
    wf[(((kt * 16 + nt) * 32 + lane) << 4) + idx] = f2bf(acc);
}

// ---------------------------------------------------------------------------
// Main kernel: one workgroup (8 wave32) per 8x8 window.
//   Phase 1: T = attn_w @ X_w     (64x64 @ 64x256, K=64)
//   Phase 2: out_w = T @ Wf + bo  (64x256 @ 256x256, K=256)
// LDS: [ XT 32KB | attnB 8KB ] during phase 1; T (32KB) aliases XT after.
// ---------------------------------------------------------------------------
__global__ __launch_bounds__(256)
void win_attn_kernel(const float* __restrict__ x,
                     const float* __restrict__ attn,
                     const float* __restrict__ bo,
                     const void* __restrict__ wf_raw,
                     float* __restrict__ out) {
    const __bf16* wf = (const __bf16*)wf_raw;

    __shared__ __attribute__((aligned(64))) unsigned char smem[40 * 1024];
    __bf16* XT    = (__bf16*)smem;            // phase 1: X^T [c][s], row-major
    __bf16* attnB = (__bf16*)(smem + 32768);  // phase 1: attn, A-fragment order
    __bf16* T     = (__bf16*)smem;            // phase 2: T, A-fragment order

    const int w  = blockIdx.x;
    const int b  = w >> 10;
    const int wh = (w >> 5) & 31;
    const int ww = w & 31;
    const size_t base = (((size_t)b * Hdim + wh * 8) * Wdim + ww * 8) * Cdim;
    const int tid = threadIdx.x;
    const float* attnw = attn + (size_t)w * (SEQ * SEQ);

    // ---- stage attn (f32, contiguous) -> bf16 A-fragment layout in LDS ----
    {
        const int row   = tid >> 2;          // token row 0..63
        const int kbase = (tid & 3) * 16;    // 16 contiguous K per thread
        const float* p  = attnw + row * SEQ + kbase;
        float v[16];
#pragma unroll
        for (int i = 0; i < 16; ++i) v[i] = p[i];
        const int kblk = kbase >> 5;                  // 32-wide K block
        const int off8 = ((kbase >> 4) & 1) * 16;     // hr = 0 vs 1 half
        __bf16* d = attnB + (kblk * SEQ + row) * 32 + off8;
        u32x4 lo = { pk2(v[0], v[1]),  pk2(v[2], v[3]),
                     pk2(v[4], v[5]),  pk2(v[6], v[7]) };   // idx 0..7 of half hr
        u32x4 hi = { pk2(v[8], v[9]),  pk2(v[10], v[11]),
                     pk2(v[12], v[13]), pk2(v[14], v[15]) }; // idx 8..15 of half hr
        *(u32x4*)d        = lo;
        *(u32x4*)(d + 8)  = hi;
    }

    // ---- gather X_w (coalesced over channels) -> transposed bf16 in LDS ----
    {
        const int c = tid; // one channel per thread
#pragma unroll
        for (int chunk = 0; chunk < 8; ++chunk) {
            float v[8];
#pragma unroll
            for (int i = 0; i < 8; ++i) {
                const int s = chunk * 8 + i;
                v[i] = x[base + (size_t)(s >> 3) * WC + (s & 7) * Cdim + c];
            }
            u32x4 o = { pk2(v[0], v[1]), pk2(v[2], v[3]),
                        pk2(v[4], v[5]), pk2(v[6], v[7]) };
            *(u32x4*)(&XT[c * SEQ + chunk * 8]) = o;
        }
    }
    __syncthreads();

    const int wv   = tid >> 5;   // wave id: owns N-tiles {2wv, 2wv+1}
    const int lane = tid & 31;
    const int r    = lane & 15;  // A row / B,C column within tile
    const int h    = lane >> 4;  // lane half

    // ---------------- Phase 1: T = attn @ X (accs stay in VGPRs) -----------
    f32x8 accT[4][2];
#pragma unroll
    for (int mt = 0; mt < 4; ++mt) {
        const bf16x16 A0 =
            *(const bf16x16*)(&attnB[(0 * SEQ + mt * 16 + r) * 32 + h * 16]);
        const bf16x16 A1 =
            *(const bf16x16*)(&attnB[(1 * SEQ + mt * 16 + r) * 32 + h * 16]);
#pragma unroll
        for (int j = 0; j < 2; ++j) {
            const int nt = wv * 2 + j;
            f32x8 acc = {};
            // B[k][n] = X[s=k][c=n] = XT[n][k]; lane holds K = 16h..16h+15
            const bf16x16 B0 =
                *(const bf16x16*)(&XT[(nt * 16 + r) * SEQ + h * 16]);
            acc = wmma_bf16(A0, B0, acc);
            const bf16x16 B1 =
                *(const bf16x16*)(&XT[(nt * 16 + r) * SEQ + 32 + h * 16]);
            acc = wmma_bf16(A1, B1, acc);
            accT[mt][j] = acc;
        }
    }
    __syncthreads(); // all waves done reading XT/attnB; safe to alias with T

    // ---- write T in A-fragment order: addr = kt*2048 + s*32 + hr*16 + idx --
#pragma unroll
    for (int mt = 0; mt < 4; ++mt) {
#pragma unroll
        for (int j = 0; j < 2; ++j) {
            const int c   = (wv * 2 + j) * 16 + r;   // output channel of T
            const int kt  = c >> 5, kin = c & 31;
            const int hr  = kin >> 4;
            const int idx = kin & 15;
            __bf16* tp = T + kt * (SEQ * 32) + hr * 16 + idx;
#pragma unroll
            for (int i = 0; i < 8; ++i)
                tp[(mt * 16 + h * 8 + i) * 32] = f2bf(accT[mt][j][i]);
        }
    }
    __syncthreads();

    // ---------------- Phase 2: out = T @ Wf + bo ---------------------------
#pragma unroll
    for (int mt = 0; mt < 4; ++mt) {
#pragma unroll
        for (int j = 0; j < 2; ++j) {
            const int nt = wv * 2 + j;
            f32x8 acc = {};
#pragma unroll
            for (int kt = 0; kt < 8; ++kt) {
                const bf16x16 Af = *(const bf16x16*)(
                    &T[(kt * SEQ + mt * 16 + r) * 32 + h * 16]);
                const bf16x16 Bf = *(const bf16x16*)(
                    wf + (((kt * 16 + nt) * 32 + lane) << 4));
                acc = wmma_bf16(Af, Bf, acc);
            }
            const int ch = nt * 16 + r;
            const float bias = bo[ch];
#pragma unroll
            for (int i = 0; i < 8; ++i) {
                const int s = mt * 16 + h * 8 + i;
                out[base + (size_t)(s >> 3) * WC + (s & 7) * Cdim + ch] =
                    acc[i] + bias;
            }
        }
    }
}

// ---------------------------------------------------------------------------
extern "C" void kernel_launch(void* const* d_in, const int* in_sizes, int n_in,
                              void* d_out, int out_size, void* d_ws, size_t ws_size,
                              hipStream_t stream) {
    const float* x    = (const float*)d_in[0]; // [4,256,256,256]
    const float* attn = (const float*)d_in[1]; // [4096,1,64,64]
    const float* Wv   = (const float*)d_in[2]; // [256,256]
    const float* Wo   = (const float*)d_in[3]; // [256,256]
    const float* bo   = (const float*)d_in[4]; // [256]
    float* out = (float*)d_out;

    // d_ws: 256*256 bf16 = 128 KB pre-swizzled fused weight Wf = Wv @ Wo
    wf_prep_kernel<<<256, 256, 0, stream>>>(Wv, Wo, d_ws);
    win_attn_kernel<<<NW, 256, 0, stream>>>(x, attn, bo, d_ws, out);
}
